// NNMFBlock3_17978733101072
// MI455X (gfx1250) — compile-verified, matrix operand order checked by reading, not run on hardware
//
#include <hip/hip_runtime.h>
#include <hip/hip_bf16.h>

// ---------------------------------------------------------------------------
// NNMF block for MI455X (gfx1250): bf16 WMMA 16x16x32, f32 accumulate.
// ~620 GFLOP of GEMM fused per 64-pixel tile; h lives in LDS for all 20
// multiplicative-update iterations. Wave grid = 2 M-pairs x 4 N-ranges;
// K-outer NT=3 x MT=2 register tile: 4 ds_load_b128 feed 6 v_wmma per
// K-chunk (LDS BW amortized) at ~130 VGPRs -- no spills.
// ---------------------------------------------------------------------------

#define DIMC    384
#define HID     768
#define NPIX    25088          // 32*28*28
#define MTILE   64
#define NWG     (NPIX / MTILE) // 392
#define NITER   20
#define EPSF    1e-6f
#define H_STR   784            // LDS row stride for h   (768 + 16 pad)
#define C_STR   400            // LDS row stride for C-dim arrays (384 + 16 pad)
#define P_STR   68             // fp32 epilogue staging stride (64 + 4 pad)
#define SMEM_BYTES 204800

typedef __attribute__((ext_vector_type(16))) __bf16 v16bf;
typedef __attribute__((ext_vector_type(8)))  float  v8f;

static __device__ __forceinline__ v16bf ld16b(const __bf16* p) {
  return *(const v16bf*)p;   // 32B aligned by construction -> b128 pair
}

// K-outer GEMM register tile: NT 16-col N-tiles x 2 M-strips, double-buffered
// A (LDS) and B (global) fragments. Each K-chunk: 4 ds_load_b128 + NT*2
// global_load_b128 feed NT*2 v_wmma_f32_16x16x32_bf16; next chunk's
// fragments are fetched while the current WMMA block executes.
template <int NT, int KT>
static __device__ __forceinline__ void gemm_mt2(
    const __bf16* __restrict__ b0,   // global B base (lane-resolved, n-tile 0)
    int bstr,                        // elements between N-tile strips (16*ldb)
    const __bf16* __restrict__ a0,   // LDS A base (lane-resolved)
    int a_str, v8f acc[NT][2]) {
  v16bf af[2], bf[NT];
  #pragma unroll
  for (int mi = 0; mi < 2; ++mi) af[mi] = ld16b(a0 + mi * 16 * a_str);
  #pragma unroll
  for (int n = 0; n < NT; ++n) bf[n] = ld16b(b0 + n * bstr);
  #pragma unroll
  for (int kc = 0; kc < KT; ++kc) {
    v16bf af_n[2], bf_n[NT];
    if (kc + 1 < KT) {
      #pragma unroll
      for (int n = 0; n < NT; ++n) {
        __builtin_prefetch(b0 + n * bstr + (kc + 2) * 32, 0, 3); // warm GL2
        bf_n[n] = ld16b(b0 + n * bstr + (kc + 1) * 32);
      }
      #pragma unroll
      for (int mi = 0; mi < 2; ++mi)
        af_n[mi] = ld16b(a0 + mi * 16 * a_str + (kc + 1) * 32);
    }
    #pragma unroll
    for (int n = 0; n < NT; ++n)
      #pragma unroll
      for (int mi = 0; mi < 2; ++mi)
        acc[n][mi] = __builtin_amdgcn_wmma_f32_16x16x32_bf16(
            false, af[mi], false, bf[n], (short)0, acc[n][mi], false, false);
    if (kc + 1 < KT) {
      #pragma unroll
      for (int mi = 0; mi < 2; ++mi) af[mi] = af_n[mi];
      #pragma unroll
      for (int n = 0; n < NT; ++n) bf[n] = bf_n[n];
    }
  }
}

template <int NT>
static __device__ __forceinline__ void zero_mt2(v8f acc[NT][2]) {
  #pragma unroll
  for (int n = 0; n < NT; ++n)
    #pragma unroll
    for (int i = 0; i < 2; ++i)
      #pragma unroll
      for (int j = 0; j < 8; ++j) acc[n][i][j] = 0.f;
}

// ---------------------------------------------------------------------------
// Kernel 1: Wp = rownorm(max(W,eps)) in bf16, plus its transpose.
// ---------------------------------------------------------------------------
__global__ __launch_bounds__(128)
void weights_kernel(const float* __restrict__ w,
                    __bf16* __restrict__ Wp, __bf16* __restrict__ WpT) {
  __shared__ float red[128];
  const int r = blockIdx.x;          // hidden row 0..767
  const int t = threadIdx.x;
  float part = 0.f;
  #pragma unroll
  for (int e = 0; e < 3; ++e) part += fmaxf(w[r * DIMC + t + e * 128], EPSF);
  red[t] = part;
  __syncthreads();
  for (int s = 64; s > 0; s >>= 1) {
    if (t < s) red[t] += red[t + s];
    __syncthreads();
  }
  const float inv = 1.0f / red[0];
  #pragma unroll
  for (int e = 0; e < 3; ++e) {
    const int c = t + e * 128;
    const float v = fmaxf(w[r * DIMC + c], EPSF) * inv;
    Wp [r * DIMC + c] = (__bf16)v;
    WpT[c * HID  + r] = (__bf16)v;
  }
}

// ---------------------------------------------------------------------------
// Kernel 2a: depthwise 7x7 conv, one block per (b,c) 28x28 plane.
// Coalesced plane load into LDS, scalar-uniform taps, NCHW output (-> d_out
// used as scratch; fully overwritten by the NNMF epilogue later).
// ---------------------------------------------------------------------------
__global__ __launch_bounds__(256)
void dwconv_kernel(const float* __restrict__ x,
                   const float* __restrict__ kern,
                   float* __restrict__ y) {
  __shared__ float sx[784];
  __shared__ float sk[49];
  const int plane = blockIdx.x;                 // b*DIMC + c
  const float* xp = x + (size_t)plane * 784;
  for (int i = threadIdx.x; i < 784; i += 256) sx[i] = xp[i];
  if (threadIdx.x < 49) sk[threadIdx.x] = kern[(plane % DIMC) * 49 + threadIdx.x];
  __syncthreads();
  for (int p = threadIdx.x; p < 784; p += 256) {
    const int hh = p / 28, ww = p % 28;
    float acc = 0.f;
    #pragma unroll
    for (int i = 0; i < 7; ++i) {
      const int hi = hh + i - 3;
      if (hi < 0 || hi >= 28) continue;
      #pragma unroll
      for (int j = 0; j < 7; ++j) {
        const int wj = ww + j - 3;
        if (wj < 0 || wj >= 28) continue;
        acc += sx[hi * 28 + wj] * sk[i * 7 + j];
      }
    }
    y[(size_t)plane * 784 + p] = acc;
  }
}

// ---------------------------------------------------------------------------
// Kernel 2b: LayerNorm + clamp + L1-normalize. 32-pixel tile transposed
// through LDS: coalesced NCHW reads, coalesced channels-last bf16 writes.
// ---------------------------------------------------------------------------
__global__ __launch_bounds__(256)
void lnnorm_kernel(const float* __restrict__ y,
                   const float* __restrict__ lnw,
                   const float* __restrict__ lnb,
                   __bf16* __restrict__ xin) {
  __shared__ float sy[32][DIMC + 4];
  __shared__ float red1[256];
  __shared__ float red2[256];
  __shared__ float smu[32], srs[32], sinv[32];
  const int tid  = threadIdx.x;
  const int pix0 = blockIdx.x * 32;

  for (int e = tid; e < 32 * DIMC; e += 256) {     // consecutive tid -> pixels
    const int c = e >> 5, p = e & 31;
    const int pix = pix0 + p;
    const int b = pix / 784, rem = pix % 784;
    sy[p][c] = y[((size_t)(b * DIMC + c)) * 784 + rem];
  }
  __syncthreads();

  const int p = tid >> 3, q = tid & 7;             // 8 threads per pixel
  float s1 = 0.f, s2 = 0.f;
  for (int c = q * 48; c < q * 48 + 48; ++c) {
    const float v = sy[p][c];
    s1 += v; s2 += v * v;
  }
  red1[tid] = s1; red2[tid] = s2;
  __syncthreads();
  if (q == 0) {
    float a = 0.f, b2 = 0.f;
    #pragma unroll
    for (int k = 0; k < 8; ++k) { a += red1[tid + k]; b2 += red2[tid + k]; }
    const float mu = a * (1.0f / DIMC);
    smu[p] = mu;
    srs[p] = rsqrtf(b2 * (1.0f / DIMC) - mu * mu + 1e-6f);
  }
  __syncthreads();
  const float mu = smu[p], rs = srs[p];
  float ss = 0.f;
  for (int c = q * 48; c < q * 48 + 48; ++c)
    ss += fmaxf((sy[p][c] - mu) * rs * lnw[c] + lnb[c], EPSF);
  red1[tid] = ss;
  __syncthreads();
  if (q == 0) {
    float a = 0.f;
    #pragma unroll
    for (int k = 0; k < 8; ++k) a += red1[tid + k];
    sinv[p] = 1.0f / a;
  }
  __syncthreads();
  for (int e = tid; e < 32 * DIMC; e += 256) {     // consecutive tid -> channels
    const int pp = e / DIMC, c = e % DIMC;
    const float v = fmaxf((sy[pp][c] - smu[pp]) * srs[pp] * lnw[c] + lnb[c], EPSF);
    xin[(size_t)(pix0 + pp) * DIMC + c] = (__bf16)(v * sinv[pp]);
  }
}

// ---------------------------------------------------------------------------
// Kernel 3: fused NNMF iterations on WMMA bf16. 256 threads = 8 wave32 laid
// out as 2 M-pairs x 4 N-ranges. C/D layout per CDNA5 ISA:
// VGPR v -> M = v (+8 for lanes 16-31), lane%16 = N.
// ---------------------------------------------------------------------------
__global__ __launch_bounds__(256)
void nnmf_kernel(const __bf16* __restrict__ xin,
                 const __bf16* __restrict__ Wp,
                 const __bf16* __restrict__ WpT,
                 const float*  __restrict__ x,
                 const float*  __restrict__ gamma,
                 float* __restrict__ out) {
  extern __shared__ char smem[];
  __bf16* sh_h     = (__bf16*)smem;                 // 64 x 784   (100,352 B)
  __bf16* sh_xin   = sh_h    + MTILE * H_STR;       // 64 x 400   ( 51,200 B)
  __bf16* sh_ratio = sh_xin  + MTILE * C_STR;       // 64 x 400   ( 51,200 B)
  float*  sh_red   = (float*)(sh_ratio + MTILE * C_STR); // 256
  float*  sh_rsum  = sh_red + 256;                  // 64
  // fp32 epilogue staging reuses dead xin/ratio/red space after sh_h:
  float*  stf      = (float*)(smem + MTILE * H_STR * 2); // 384 x 68 (104,448 B)

  const int tid    = threadIdx.x;
  const int lane   = tid & 31;
  const int wid    = tid >> 5;
  const int mpair  = wid >> 2;           // 0..1 : which pair of 16-row strips
  const int nrange = wid & 3;            // 0..3 : which quarter of N
  const int pix0   = blockIdx.x * MTILE;
  const int mrow   = lane & 15;          // A/B fragment row within tile
  const int khalf  = (lane >> 4) * 16;   // K-half select per lane group
  const int rhalf  = (lane >> 4) * 8;    // C/D row offset for lanes 16-31
  const int mbase  = mpair * 32;         // first M row owned by this wave

  // --- stage xin tile (bf16, 16B vectors), init h = 1/HID ------------------
  for (int idx = tid; idx < MTILE * 48; idx += 256) {
    const int r = idx / 48, q = idx % 48;
    ((uint4*)sh_xin)[r * (C_STR / 8) + q] =
        ((const uint4*)xin)[(size_t)(pix0 + r) * 48 + q];
  }
  {
    const unsigned short hb =
        __builtin_bit_cast(unsigned short, (__bf16)(1.0f / (float)HID));
    const unsigned int hpack = (unsigned int)hb | ((unsigned int)hb << 16);
    unsigned int* hp = (unsigned int*)sh_h;
    for (int i = tid; i < MTILE * H_STR / 2; i += 256) hp[i] = hpack;
  }
  __syncthreads();

  for (int it = 0; it < NITER; ++it) {
    // ---- GEMM1: recon = h(64x768) @ Wp(768x384); ratio = xin/max(recon,eps)
    // Wave covers 6 N-tiles (nrange*6 ..) x 2 M-strips, as two NT=3 groups.
    for (int g = 0; g < 2; ++g) {
      const int nt0 = nrange * 6 + g * 3;
      v8f acc[3][2];
      zero_mt2<3>(acc);
      gemm_mt2<3, HID / 32>(
          WpT + (size_t)(nt0 * 16 + mrow) * HID + khalf, 16 * HID,
          sh_h + (mbase + mrow) * H_STR + khalf, H_STR, acc);
      #pragma unroll
      for (int n = 0; n < 3; ++n) {
        const int col = (nt0 + n) * 16 + (lane & 15);
        #pragma unroll
        for (int mi = 0; mi < 2; ++mi)
          #pragma unroll
          for (int v = 0; v < 8; ++v) {
            const int m = mbase + mi * 16 + v + rhalf;
            const float xv = (float)sh_xin[m * C_STR + col];
            sh_ratio[m * C_STR + col] =
                (__bf16)(xv / fmaxf(acc[n][mi][v], EPSF));
          }
      }
    }
    __syncthreads();

    // ---- GEMM2: hu = ratio(64x384) @ WpT(384x768);  h *= hu  --------------
    // Wave covers 12 N-tiles (nrange*12 ..) x 2 M-strips, as four NT=3 groups.
    for (int g = 0; g < 4; ++g) {
      const int nt0 = nrange * 12 + g * 3;
      v8f acc[3][2];
      zero_mt2<3>(acc);
      gemm_mt2<3, DIMC / 32>(
          Wp + (size_t)(nt0 * 16 + mrow) * DIMC + khalf, 16 * DIMC,
          sh_ratio + (mbase + mrow) * C_STR + khalf, C_STR, acc);
      #pragma unroll
      for (int n = 0; n < 3; ++n) {
        const int col = (nt0 + n) * 16 + (lane & 15);
        #pragma unroll
        for (int mi = 0; mi < 2; ++mi)
          #pragma unroll
          for (int v = 0; v < 8; ++v) {
            const int m  = mbase + mi * 16 + v + rhalf;
            const int hi = m * H_STR + col;
            sh_h[hi] = (__bf16)((float)sh_h[hi] * acc[n][mi][v]);
          }
      }
    }
    __syncthreads();

    // ---- renormalize rows of h -------------------------------------------
    {
      const int row = tid >> 2, part = tid & 3;
      float s = 0.f;
      const __bf16* hr = sh_h + row * H_STR + part * 192;
      for (int j = 0; j < 192; ++j) s += (float)hr[j];
      sh_red[tid] = s;
    }
    __syncthreads();
    if (tid < MTILE) {
      const float tot = sh_red[tid * 4] + sh_red[tid * 4 + 1] +
                        sh_red[tid * 4 + 2] + sh_red[tid * 4 + 3];
      sh_rsum[tid] = 1.0f / tot;
    }
    __syncthreads();
    for (int e = tid; e < MTILE * HID; e += 256) {
      const int r = e / HID, c = e % HID;
      const int hi = r * H_STR + c;
      sh_h[hi] = (__bf16)((float)sh_h[hi] * sh_rsum[r]);
    }
    __syncthreads();
  }

  // ---- final recon, layer-scale; stage fp32 [c][p] tile in LDS ------------
  for (int g = 0; g < 2; ++g) {
    const int nt0 = nrange * 6 + g * 3;
    v8f acc[3][2];
    zero_mt2<3>(acc);
    gemm_mt2<3, HID / 32>(
        WpT + (size_t)(nt0 * 16 + mrow) * HID + khalf, 16 * HID,
        sh_h + (mbase + mrow) * H_STR + khalf, H_STR, acc);
    #pragma unroll
    for (int n = 0; n < 3; ++n) {
      const int c = (nt0 + n) * 16 + (lane & 15);
      const float gsc = gamma[c];
      #pragma unroll
      for (int mi = 0; mi < 2; ++mi)
        #pragma unroll
        for (int v = 0; v < 8; ++v) {
          const int m = mbase + mi * 16 + v + rhalf;
          stf[c * P_STR + m] = gsc * acc[n][mi][v];  // stf does not alias sh_h
        }
    }
  }
  __syncthreads();

  // ---- coalesced float4 residual-add NCHW store ---------------------------
  // 784 % 4 == 0, so a 4-pixel run never crosses an image boundary.
  for (int e = tid; e < DIMC * 16; e += 256) {
    const int c = e / 16;
    const int p = (e % 16) * 4;
    const int pix = pix0 + p;
    const int b = pix / 784, rem = pix % 784;
    const size_t idx = ((size_t)(b * DIMC + c)) * 784 + rem;
    const float4 rv = *(const float4*)(stf + c * P_STR + p);
    const float4 xv = *(const float4*)(x + idx);
    float4 o;
    o.x = xv.x + rv.x; o.y = xv.y + rv.y;
    o.z = xv.z + rv.z; o.w = xv.w + rv.w;
    *(float4*)(out + idx) = o;
  }
}

// ---------------------------------------------------------------------------
extern "C" void kernel_launch(void* const* d_in, const int* in_sizes, int n_in,
                              void* d_out, int out_size, void* d_ws, size_t ws_size,
                              hipStream_t stream) {
  (void)in_sizes; (void)out_size; (void)ws_size;
  if (n_in < 6) return;
  const float* x    = (const float*)d_in[0];   // (32,384,28,28)
  const float* kern = (const float*)d_in[1];   // (384,1,7,7)
  const float* lnw  = (const float*)d_in[2];   // (384,)
  const float* lnb  = (const float*)d_in[3];   // (384,)
  const float* w    = (const float*)d_in[4];   // (768,384)
  const float* gam  = (const float*)d_in[5];   // (384,)
  float* out = (float*)d_out;

  char* ws = (char*)d_ws;
  __bf16* xin = (__bf16*)(ws);                                     // 19.3 MB
  __bf16* Wp  = (__bf16*)(ws + (size_t)NPIX * DIMC * 2);           // 0.59 MB
  __bf16* WpT = (__bf16*)(ws + (size_t)NPIX * DIMC * 2 + (size_t)HID * DIMC * 2);

  // d_out doubles as NCHW scratch for the conv output y; it is fully
  // overwritten by nnmf_kernel's epilogue afterwards (same stream order).
  float* y = out;

  weights_kernel  <<<HID, 128, 0, stream>>>(w, Wp, WpT);
  dwconv_kernel   <<<32 * DIMC, 256, 0, stream>>>(x, kern, y);
  lnnorm_kernel   <<<NPIX / 32, 256, 0, stream>>>(y, lnw, lnb, xin);
  nnmf_kernel     <<<NWG, 256, SMEM_BYTES, stream>>>(xin, Wp, WpT, x, gam, out);
}
